// Network_30245159698513
// MI455X (gfx1250) — matrix operand order repaired and checked by reading, not compiled, not run on hardware
//
#include <hip/hip_runtime.h>
#include <math.h>

#define KNB 16
#define BN_COLS 256          // columns staged per block (4 waves x 4 tiles x 16)
#define BRS 48               // LDS row stride in halfs (96B: 32B-aligned rows)

typedef __attribute__((ext_vector_type(16))) _Float16 v16h;
typedef __attribute__((ext_vector_type(2)))  _Float16 v2h;
typedef __attribute__((ext_vector_type(8)))  float    v8f;

// ---------------------------------------------------------------------------
// Weight pre-pack: W [Cout,Cin] f32 -> f16 fragments in a contiguous-k (k = 16*half + e) WMMA
// A-matrix (16x32 f16) VGPR layout, zero padded.
//   Wp[(((ct*nks)+ks)*32 + lane)*16 + e]
//   half = lane>>4, l16 = lane&15, row = ct*16+l16
//   koff = e + half*16 ;  k = ks*32 + koff
// ---------------------------------------------------------------------------
__global__ void pack_w_k(const float* __restrict__ W, _Float16* __restrict__ Wp,
                         int Cout, int Cin, int nks, long total)
{
  long t = (long)blockIdx.x * blockDim.x + threadIdx.x;
  if (t >= total) return;
  int e    = (int)(t & 15);
  int lane = (int)((t >> 4) & 31);
  long fs  = t >> 9;                 // ct*nks + ks
  int ks   = (int)(fs % nks);
  int ct   = (int)(fs / nks);
  int half = lane >> 4, l16 = lane & 15;
  int row  = ct * 16 + l16;
  int koff = e + half * 16;
  int k    = ks * 32 + koff;
  float v = (row < Cout && k < Cin) ? W[(size_t)row * Cin + k] : 0.0f;
  Wp[t] = (_Float16)v;
}

// ---------------------------------------------------------------------------
// 1x1-conv + folded-BN + optional ReLU + optional residual, via WMMA.
//   Y[b,co,m] = relu( G[co] * sum_ci W[co,ci]*X[b,ci,m] + Bs[co] ) (+ Res)
// Virtual channel concat: ci < C1 reads X1, else X2[ci-C1].
// Block = 128 threads (4 waves) = 16(co) x 256(m) tile of Y.
// Per K-step: the block stages a 32(k) x 256(n) X tile into LDS as f16
// ([n][k] layout, 96B row stride), block-uniform concat/bounds handling,
// zero-padded channels, clamped tail columns.  The compute loop is fully
// BRANCHLESS (all 4 tiles computed unconditionally; out-of-range tiles are
// harmless garbage on clamped data) -- validity enforced only at the stores.
// REQUIRES M % 16 == 0 (all call sites).
// ---------------------------------------------------------------------------
__global__ __launch_bounds__(128) void conv1x1_wmma(
    const float* __restrict__ X1, int C1,
    const float* __restrict__ X2, int C2,
    const _Float16* __restrict__ Wp,
    const float* __restrict__ Gs,
    const float* __restrict__ Bs,
    const float* __restrict__ Res,
    float* __restrict__ Y,
    int Cout, int M, int nks, int doRelu)
{
  __shared__ _Float16 sB[BN_COLS * BRS];          // 24 KB staged B tile
  const int Cin = C1 + C2;
  const int tid  = threadIdx.x;
  const int lane = tid & 31;
  const int wave = tid >> 5;
  const int ct  = blockIdx.y;
  const int co0 = ct * 16;
  const int bz  = blockIdx.z;

  const size_t x1off = (size_t)bz * (size_t)C1 * M;
  const size_t x2off = (size_t)bz * (size_t)C2 * M;
  const size_t yoff  = (size_t)bz * (size_t)Cout * M;

  const int half = lane >> 4;
  const int l16  = lane & 15;

  const int nblk = blockIdx.x * BN_COLS;
  const int m0   = nblk + wave * 64;
  const int tv   = (m0 < M) ? min(4, (M - m0) >> 4) : 0;   // wave-uniform
  const bool tail = (nblk + BN_COLS) > M;                  // block-uniform

  const int n0  = 2 * tid;                                 // local column pair
  const int gc0 = nblk + n0;                               // global column

  v8f acc[4] = {{}, {}, {}, {}};
  const _Float16* wbase = Wp + ((size_t)ct * nks * 32 + lane) * 16;
  const int ncol0 = wave * 64 + l16;                       // tile-0 local col

  for (int ks = 0; ks < nks; ++ks) {
    __syncthreads();                 // previous fragment reads done
    // ---- stage 32 x 256 tile: sB[n][k] (f16) ----
    for (int kp = 0; kp < 16; ++kp) {
      const int k  = 2 * kp;
      const int c0 = ks * 32 + k, c1 = c0 + 1;
      float x00 = 0.f, x01 = 0.f, x10 = 0.f, x11 = 0.f;
      if (!tail) {
        if (c0 < Cin) {                                  // block-uniform
          const float* s = (c0 < C1) ? X1 + x1off + (size_t)c0 * M
                                     : X2 + x2off + (size_t)(c0 - C1) * M;
          x00 = s[gc0]; x10 = s[gc0 + 1];                // merges to b64
        }
        if (c1 < Cin) {
          const float* s = (c1 < C1) ? X1 + x1off + (size_t)c1 * M
                                     : X2 + x2off + (size_t)(c1 - C1) * M;
          x01 = s[gc0]; x11 = s[gc0 + 1];
        }
      } else {                                           // tail block: clamp
        const int g0 = min(gc0, M - 1), g1 = min(gc0 + 1, M - 1);
        if (c0 < Cin) {
          const float* s = (c0 < C1) ? X1 + x1off + (size_t)c0 * M
                                     : X2 + x2off + (size_t)(c0 - C1) * M;
          x00 = s[g0]; x10 = s[g1];
        }
        if (c1 < Cin) {
          const float* s = (c1 < C1) ? X1 + x1off + (size_t)c1 * M
                                     : X2 + x2off + (size_t)(c1 - C1) * M;
          x01 = s[g0]; x11 = s[g1];
        }
      }
      v2h p0 = {(_Float16)x00, (_Float16)x01};
      v2h p1 = {(_Float16)x10, (_Float16)x11};
      *(v2h*)(&sB[n0 * BRS + k])       = p0;
      *(v2h*)(&sB[(n0 + 1) * BRS + k]) = p1;
    }
    __syncthreads();

    // ---- compute: 1 A fragment -> 4 WMMAs, fully branchless ----
    v16h a = *(const v16h*)(wbase + (size_t)ks * 512);
#pragma unroll
    for (int t = 0; t < 4; ++t) {
      v16h bm = *(const v16h*)(&sB[(ncol0 + t * 16) * BRS + half * 16]);
      acc[t] = __builtin_amdgcn_wmma_f32_16x16x32_f16(
          false, a, false, bm, (short)0, acc[t], false, false);
    }
  }

#pragma unroll
  for (int t = 0; t < 4; ++t) {
    if (t < tv) {
      const int n = m0 + t * 16 + l16;
#pragma unroll
      for (int r = 0; r < 8; ++r) {
        int co = co0 + r + 8 * half;
        if (co < Cout) {
          float y = acc[t][r] * Gs[co] + Bs[co];
          if (doRelu) y = fmaxf(y, 0.0f);
          size_t idx = yoff + (size_t)co * M + n;
          if (Res) y += Res[idx];
          Y[idx] = y;
        }
      }
    }
  }
}

// ---------------------------------------------------------------------------
// Neighbor gather: out[b,c,n,k] = src[b*srcBatch + c*sC + idx[b,n,k]*sN]
// ---------------------------------------------------------------------------
__global__ void gather_nb_k(const float* __restrict__ src,
                            const int* __restrict__ idx,
                            float* __restrict__ out,
                            int C, int Nd, int Ks, int sC, int sN,
                            long srcBatch)
{
  const int b = blockIdx.y;
  const long NK = (long)Nd * Ks;
  long t = (long)blockIdx.x * blockDim.x + threadIdx.x;
  if (t >= NK) return;
  int j = idx[(size_t)b * NK + t];
  const float* s = src + (size_t)b * srcBatch + (size_t)j * sN;
  float* o = out + (size_t)b * C * NK + t;
  for (int c = 0; c < C; ++c) o[(size_t)c * NK] = s[(size_t)c * sC];
}

// fi = concat(nf - tf, tf);  f:[B,dh,N], nf:[B,dh,N,K] -> fi:[B,2dh,N,K]
__global__ void build_fi_k(const float* __restrict__ nf,
                           const float* __restrict__ f,
                           float* __restrict__ fi,
                           int dh, int N, int Ks)
{
  const int b = blockIdx.y;
  const long NK = (long)N * Ks;
  long t = (long)blockIdx.x * blockDim.x + threadIdx.x;
  if (t >= NK) return;
  const int n = (int)(t / Ks);
  const float* fp  = f  + (size_t)b * dh * N;
  const float* nfp = nf + (size_t)b * dh * NK;
  float* fo = fi + (size_t)b * 2 * dh * NK;
  for (int c = 0; c < dh; ++c) {
    float fv = fp[(size_t)c * N + n];
    float nv = nfp[(size_t)c * NK + t];
    fo[(size_t)c * NK + t] = nv - fv;
    fo[(size_t)(c + dh) * NK + t] = fv;
  }
}

// xi = concat(nx - tx, nx + d5, tx);  nx,d5:[B,3,N,K], xyz:[B,N,3] -> xi:[B,9,N,K]
__global__ void build_xi_k(const float* __restrict__ nx,
                           const float* __restrict__ d5,
                           const float* __restrict__ xyz,
                           float* __restrict__ xi,
                           int N, int Ks)
{
  const int b = blockIdx.y;
  const long NK = (long)N * Ks;
  long t = (long)blockIdx.x * blockDim.x + threadIdx.x;
  if (t >= NK) return;
  const int n = (int)(t / Ks);
  const float* nxp = nx + (size_t)b * 3 * NK;
  const float* dp  = d5 + (size_t)b * 3 * NK;
  const float* xp  = xyz + ((size_t)b * N + n) * 3;
  float* xo = xi + (size_t)b * 9 * NK;
  for (int c = 0; c < 3; ++c) {
    float tx = xp[c];
    float nv = nxp[(size_t)c * NK + t];
    xo[(size_t)c * NK + t]       = nv - tx;
    xo[(size_t)(3 + c) * NK + t] = nv + dp[(size_t)c * NK + t];
    xo[(size_t)(6 + c) * NK + t] = tx;
  }
}

// in-place softmax over trailing K on [B,C,N,K]
__global__ void softmax_k_k(float* __restrict__ x, int C, int N, int Ks)
{
  const int b = blockIdx.y;
  const long CN = (long)C * N;
  long t = (long)blockIdx.x * blockDim.x + threadIdx.x;
  if (t >= CN) return;
  float* p = x + ((size_t)b * CN + t) * Ks;
  float m = -INFINITY;
  for (int k = 0; k < Ks; ++k) m = fmaxf(m, p[k]);
  float s = 0.0f;
  for (int k = 0; k < Ks; ++k) { float e = expf(p[k] - m); p[k] = e; s += e; }
  float inv = 1.0f / s;
  for (int k = 0; k < Ks; ++k) p[k] *= inv;
}

// attention pool: ov = concat(xe,fe) [do=2dh ch]; P:[B,2*do,N] = [max_k ov ; sum_k ov*kw]
__global__ void pool_attn_k(const float* __restrict__ xe,
                            const float* __restrict__ fe,
                            const float* __restrict__ kw,
                            float* __restrict__ P,
                            int dh, int N, int Ks)
{
  const int dO = 2 * dh;
  const int b = blockIdx.y;
  const long CN = (long)dO * N;
  long t = (long)blockIdx.x * blockDim.x + threadIdx.x;
  if (t >= CN) return;
  const int c = (int)(t / N), n = (int)(t % N);
  const float* ov = ((c < dh) ? (xe + ((size_t)b * dh + c) * N * Ks)
                              : (fe + ((size_t)b * dh + (c - dh)) * N * Ks))
                    + (size_t)n * Ks;
  const float* kp = kw + ((size_t)b * dO + c) * N * Ks + (size_t)n * Ks;
  float mx = -INFINITY, ws = 0.0f;
  for (int k = 0; k < Ks; ++k) { float v = ov[k]; mx = fmaxf(mx, v); ws += v * kp[k]; }
  P[((size_t)b * 2 * dO + c) * N + n]      = mx;
  P[((size_t)b * 2 * dO + dO + c) * N + n] = ws;
}

// max-pool gather: out[b,c,nd] = max_k f[b,c,sidx[b,nd,k]]
__global__ void sample_max_k(const float* __restrict__ f,
                             const int* __restrict__ sidx,
                             float* __restrict__ out,
                             int C, int Nd, int Ns, int Ks)
{
  const int b = blockIdx.y;
  const long CN = (long)C * Nd;
  long t = (long)blockIdx.x * blockDim.x + threadIdx.x;
  if (t >= CN) return;
  const int c = (int)(t / Nd), nd = (int)(t % Nd);
  const int* ip = sidx + ((size_t)b * Nd + nd) * Ks;
  const float* fp = f + ((size_t)b * C + c) * Ns;
  float m = -INFINITY;
  for (int k = 0; k < Ks; ++k) m = fmaxf(m, fp[ip[k]]);
  out[((size_t)b * C + c) * Nd + nd] = m;
}

// nearest upsample: out[b,c,n] = f[b,c,iidx[b,n]]
__global__ void interp_k(const float* __restrict__ f,
                         const int* __restrict__ iidx,
                         float* __restrict__ out,
                         int C, int N, int Ns)
{
  const int b = blockIdx.y;
  const long CN = (long)C * N;
  long t = (long)blockIdx.x * blockDim.x + threadIdx.x;
  if (t >= CN) return;
  const int c = (int)(t / N), n = (int)(t % N);
  int j = iidx[(size_t)b * N + n];
  out[((size_t)b * C + c) * N + n] = f[((size_t)b * C + c) * Ns + j];
}

// softmax over 4 scale logits, blend f_multi
__global__ void wsum_final_k(const float* __restrict__ f0, const float* __restrict__ f1,
                             const float* __restrict__ f2, const float* __restrict__ f3,
                             const float* __restrict__ w0, const float* __restrict__ w1,
                             const float* __restrict__ w2, const float* __restrict__ w3,
                             float* __restrict__ out, int C, int N)
{
  const int b = blockIdx.y;
  long n = (long)blockIdx.x * blockDim.x + threadIdx.x;
  if (n >= N) return;
  float l0 = w0[(size_t)b * N + n], l1 = w1[(size_t)b * N + n];
  float l2 = w2[(size_t)b * N + n], l3 = w3[(size_t)b * N + n];
  float m = fmaxf(fmaxf(l0, l1), fmaxf(l2, l3));
  float e0 = expf(l0 - m), e1 = expf(l1 - m), e2 = expf(l2 - m), e3 = expf(l3 - m);
  float inv = 1.0f / (e0 + e1 + e2 + e3);
  for (int c = 0; c < C; ++c) {
    size_t o = ((size_t)b * C + c) * N + n;
    out[o] = (e0 * f0[o] + e1 * f1[o] + e2 * f2[o] + e3 * f3[o]) * inv;
  }
}

// ---------------------------------------------------------------------------
extern "C" void kernel_launch(void* const* d_in, const int* in_sizes, int n_in,
                              void* d_out, int out_size, void* d_ws, size_t ws_size,
                              hipStream_t stream)
{
  (void)in_sizes; (void)n_in; (void)out_size; (void)ws_size;
  const int NSL[5]  = {40960, 10240, 2560, 640, 160};
  const int DOUT[4] = {16, 64, 128, 256};
  const int DIMS[5] = {32, 32, 128, 256, 512};
  const int DEC_IDX[4] = {0, 4, 7, 9};
  const int Bn = 2;

  const float* features = (const float*)d_in[0];
  const float* xyz[4]; const int* nidx[4]; const int* sidx[4]; const int* iidx[4];
  for (int i = 0; i < 4; ++i) {
    xyz[i]  = (const float*)d_in[1 + 4 * i];
    nidx[i] = (const int*)  d_in[2 + 4 * i];
    sidx[i] = (const int*)  d_in[3 + 4 * i];
    iidx[i] = (const int*)  d_in[4 + 4 * i];
  }
  struct PW { const float *W, *g, *b; };
  auto getp = [&](int base) {
    PW p; p.W = (const float*)d_in[base]; p.g = (const float*)d_in[base + 1];
    p.b = (const float*)d_in[base + 2]; return p;
  };
  // params flattened in setup_inputs insertion order: fc0, enc[i]{mlp1,5,6,7,8,9,10,11},
  // dec[0..3], dec2[0..9], wb[0..3], fc1, fc2, fc3 ; each {W,g,b}
  PW fc0p = getp(17);
  PW encp[4][8];
  for (int i = 0; i < 4; ++i)
    for (int m = 0; m < 8; ++m) encp[i][m] = getp(20 + 24 * i + 3 * m);
  PW decp[4];  for (int n = 0; n < 4; ++n)  decp[n]  = getp(116 + 3 * n);
  PW dec2p[10];for (int j = 0; j < 10; ++j) dec2p[j] = getp(128 + 3 * j);
  PW wbp[4];   for (int n = 0; n < 4; ++n)  wbp[n]   = getp(158 + 3 * n);
  PW fc1p = getp(170), fc2p = getp(173), fc3p = getp(176);

  // bump allocator over d_ws (stream-ordered reuse is safe)
  char* wsbase = (char*)d_ws;
  size_t wsoff = 0;
  auto allocb = [&](size_t bytes) -> void* {
    void* p = (void*)(wsbase + wsoff);
    wsoff += (bytes + 255) & ~(size_t)255;
    return p;
  };
  auto alloc = [&](size_t nf) -> float* { return (float*)allocb(nf * sizeof(float)); };

  auto conv = [&](const float* X1, int C1, const float* X2, int C2, PW p,
                  const float* Res, float* Y, int Cout, int M, int relu) {
    int Cin = C1 + C2;
    int nks = (Cin + 31) / 32;
    int nct = (Cout + 15) / 16;
    long wtotal = (long)nct * nks * 512;
    _Float16* Wp = (_Float16*)allocb((size_t)wtotal * sizeof(_Float16));
    pack_w_k<<<dim3((unsigned)((wtotal + 255) / 256)), 256, 0, stream>>>(
        p.W, Wp, Cout, Cin, nks, wtotal);
    dim3 g((unsigned)((M + BN_COLS - 1) / BN_COLS), nct, Bn);
    conv1x1_wmma<<<g, 128, 0, stream>>>(X1, C1, X2, C2, Wp, p.g, p.b, Res, Y,
                                        Cout, M, nks, relu);
  };
  auto ew = [&](long work) { return dim3((unsigned)((work + 255) / 256), Bn); };

  // ---- fc0 : 3 -> 8 over N0 ----
  float* feat0 = alloc((size_t)Bn * 8 * NSL[0]);
  conv(features, 3, nullptr, 0, fc0p, nullptr, feat0, 8, NSL[0], 1);

  // ---- persistent encoder outputs (enc_list) ----
  float* e[5];
  for (int l = 0; l < 5; ++l) e[l] = alloc((size_t)Bn * DIMS[l] * NSL[l]);

  // ---- encoder ----
  const float* cur = feat0;
  int din = 8;
  for (int i = 0; i < 4; ++i) {
    const int dO = DOUT[i], dh = dO / 2, N = NSL[i];
    const long NK = (long)N * KNB;
    const size_t mark = wsoff;

    float* f = alloc((size_t)Bn * dh * N);
    conv(cur, din, nullptr, 0, encp[i][0], nullptr, f, dh, N, 1);          // mlp1

    float* nf = alloc((size_t)Bn * dh * NK);
    gather_nb_k<<<ew(NK), 256, 0, stream>>>(f, nidx[i], nf, dh, N, KNB, N, 1,
                                            (long)dh * N);
    float* nx = alloc((size_t)Bn * 3 * NK);
    gather_nb_k<<<ew(NK), 256, 0, stream>>>(xyz[i], nidx[i], nx, 3, N, KNB, 1, 3,
                                            (long)3 * N);
    float* fi = alloc((size_t)Bn * dO * NK);
    build_fi_k<<<ew(NK), 256, 0, stream>>>(nf, f, fi, dh, N, KNB);

    float* d5 = alloc((size_t)Bn * 3 * NK);
    conv(fi, dO, nullptr, 0, encp[i][1], nullptr, d5, 3, (int)NK, 1);      // mlp5

    float* xi = alloc((size_t)Bn * 9 * NK);
    build_xi_k<<<ew(NK), 256, 0, stream>>>(nx, d5, xyz[i], xi, N, KNB);

    float* sf = alloc((size_t)Bn * dh * NK);
    conv(xi, 9, nullptr, 0, encp[i][2], nf, sf, dh, (int)NK, 1);           // mlp6 + nf
    float* xe = alloc((size_t)Bn * dh * NK);
    conv(xi, 9, nullptr, 0, encp[i][3], nullptr, xe, dh, (int)NK, 1);      // mlp7
    float* fe = alloc((size_t)Bn * dh * NK);
    conv(sf, dh, fi, dO, encp[i][4], nullptr, fe, dh, (int)NK, 1);         // mlp8(concat)

    float* kw = alloc((size_t)Bn * dO * NK);
    conv(xe, dh, fe, dh, encp[i][5], nullptr, kw, dO, (int)NK, 0);         // mlp9(concat)
    softmax_k_k<<<ew((long)dO * N), 256, 0, stream>>>(kw, dO, N, KNB);

    float* P = alloc((size_t)Bn * 2 * dO * N);
    pool_attn_k<<<ew((long)dO * N), 256, 0, stream>>>(xe, fe, kw, P, dh, N, KNB);

    float* emb = alloc((size_t)Bn * dO * N);
    conv(P, 2 * dO, nullptr, 0, encp[i][6], nullptr, emb, dO, N, 1);       // mlp10
    float* feo = (i == 0) ? e[0] : alloc((size_t)Bn * 2 * dO * N);
    conv(emb, dO, nullptr, 0, encp[i][7], nullptr, feo, 2 * dO, N, 1);     // mlp11

    sample_max_k<<<ew((long)2 * dO * NSL[i + 1]), 256, 0, stream>>>(
        feo, sidx[i], e[i + 1], 2 * dO, NSL[i + 1], N, KNB);

    cur = e[i + 1];
    din = 2 * dO;
    wsoff = mark;   // reclaim per-layer scratch (stream-ordered reuse is safe)
  }

  // ---- decoder ----
  float* fm[4]; float* fwl[4];
  for (int n = 0; n < 4; ++n) {
    int lvl = 4 - n;
    int Cf = DIMS[lvl];
    float* feature = alloc((size_t)Bn * Cf * NSL[lvl]);
    conv(e[lvl], Cf, nullptr, 0, decp[n], nullptr, feature, Cf, NSL[lvl], 1);
    for (int j = 0; j < 4 - n; ++j) {
      int Ld = 3 - n - j;                 // destination level
      float* fint = alloc((size_t)Bn * Cf * NSL[Ld]);
      interp_k<<<ew((long)Cf * NSL[Ld]), 256, 0, stream>>>(
          feature, iidx[Ld], fint, Cf, NSL[Ld], NSL[Ld + 1]);
      int Ce = DIMS[Ld];
      float* fd = alloc((size_t)Bn * Ce * NSL[Ld]);
      conv(e[Ld], Ce, fint, Cf, dec2p[DEC_IDX[n] + j], nullptr, fd, Ce, NSL[Ld], 1);
      feature = fd;
      Cf = Ce;
    }
    fm[n] = feature;                                              // [B,32,N0]
    fwl[n] = alloc((size_t)Bn * NSL[0]);
    conv(feature, Cf, nullptr, 0, wbp[n], nullptr, fwl[n], 1, NSL[0], 0);
  }

  // ---- multi-scale blend + classifier head ----
  float* finalb = alloc((size_t)Bn * 32 * NSL[0]);
  wsum_final_k<<<ew(NSL[0]), 256, 0, stream>>>(fm[0], fm[1], fm[2], fm[3],
                                               fwl[0], fwl[1], fwl[2], fwl[3],
                                               finalb, 32, NSL[0]);
  float* x1 = alloc((size_t)Bn * 64 * NSL[0]);
  conv(finalb, 32, nullptr, 0, fc1p, nullptr, x1, 64, NSL[0], 1);
  float* x2 = alloc((size_t)Bn * 32 * NSL[0]);
  conv(x1, 64, nullptr, 0, fc2p, nullptr, x2, 32, NSL[0], 1);
  conv(x2, 32, nullptr, 0, fc3p, nullptr, (float*)d_out, 13, NSL[0], 0);
}
